// TransformerBlock_65455301591688
// MI455X (gfx1250) — matrix operand order repaired and checked by reading, not compiled, not run on hardware
//
#include <hip/hip_runtime.h>
#include <hip/hip_bf16.h>

typedef __attribute__((ext_vector_type(16))) _Float16 v16h;
typedef __attribute__((ext_vector_type(8)))  _Float16 v8h;
typedef __attribute__((ext_vector_type(8)))  float    v8f;
typedef __attribute__((ext_vector_type(4)))  int      v4i;

#define AS1 __attribute__((address_space(1)))
#define AS3 __attribute__((address_space(3)))

#if defined(__has_builtin)
#if __has_builtin(__builtin_amdgcn_global_load_async_to_lds_b128)
#define HAVE_ASYNC_LDS 1
#endif
#endif
#ifndef HAVE_ASYNC_LDS
#define HAVE_ASYNC_LDS 0
#endif

// ---------------------------------------------------------------------------
// 16-byte global -> LDS stage: async-copy path (ASYNCcnt) if available
// ---------------------------------------------------------------------------
__device__ __forceinline__ void stage16B(const _Float16* g, _Float16* l) {
#if HAVE_ASYNC_LDS
  __builtin_amdgcn_global_load_async_to_lds_b128((AS1 v4i*)g, (AS3 v4i*)l, 0,
                                                 0);
#else
  *(float4*)l = *(const float4*)g;
#endif
}

__device__ __forceinline__ void async_fence() {
#if HAVE_ASYNC_LDS
#if __has_builtin(__builtin_amdgcn_s_wait_asynccnt)
  __builtin_amdgcn_s_wait_asynccnt(0);
#else
  asm volatile("s_wait_asynccnt 0x0" ::: "memory");
#endif
#endif
}

// ---------------------------------------------------------------------------
// WMMA wrapper (CDNA5 wave32, f16 in / f32 accum)
// ---------------------------------------------------------------------------
__device__ __forceinline__ v8f wmma_f16(v16h a, v16h b, v8f c) {
  return __builtin_amdgcn_wmma_f32_16x16x32_f16(false, a, false, b, (short)0,
                                                c, false, false);
}

// Fragment load from swizzled LDS: [group(512 halves)][lane][16 halves]
__device__ __forceinline__ v16h ld_frag(const _Float16* s, int group,
                                        int lane) {
  return *(const v16h*)&s[group * 512 + lane * 16];
}

// Swizzled A-operand slot for element (m, k) with 32-wide K groups (K<=64).
// Matches ISA 7.12.2 "16-bit A-Matrix 16x32" lane layout.
__device__ __forceinline__ int aslot64(int m, int k) {
  const int km = k & 31;
  return ((m >> 4) * 2 + (k >> 5)) * 512 +
         ((m & 15) + 16 * ((km >> 3) & 1)) * 16 + (km & 7) + 8 * (km >> 4);
}

// Swizzled C/D (f32 16x16) slot for element (row r, col n), 64x64 tile:
// [(r>>4)*4 + (n>>4)][lane = (n&15)+16*((r>>3)&1)][j = r&7]
__device__ __forceinline__ int cslot(int r, int n) {
  return (((r >> 4) * 4 + (n >> 4)) * 32 + (n & 15) + 16 * ((r >> 3) & 1)) * 8 +
         (r & 7);
}

// ---------------------------------------------------------------------------
// f32 -> f16 weight conversion
// ---------------------------------------------------------------------------
__global__ void cvt_f32_to_f16(const float* __restrict__ in,
                               _Float16* __restrict__ out, int n) {
  int i = blockIdx.x * 256 + threadIdx.x;
  if (i < n) out[i] = (_Float16)in[i];
}

// ---------------------------------------------------------------------------
// LayerNorm (C=1024) -> f16, one row per block, 256 threads
// ---------------------------------------------------------------------------
__global__ __launch_bounds__(256) void ln_f16(const float* __restrict__ x,
                                              const float* __restrict__ g,
                                              const float* __restrict__ bta,
                                              _Float16* __restrict__ out) {
  constexpr int C = 1024;
  __shared__ float red[256];
  const int t = threadIdx.x;
  const size_t row = blockIdx.x;
  const float4 xv = ((const float4*)(x + row * C))[t];

  red[t] = xv.x + xv.y + xv.z + xv.w;
  __syncthreads();
  for (int o = 128; o > 0; o >>= 1) {
    if (t < o) red[t] += red[t + o];
    __syncthreads();
  }
  const float mu = red[0] * (1.0f / C);
  __syncthreads();

  const float d0 = xv.x - mu, d1 = xv.y - mu, d2 = xv.z - mu, d3 = xv.w - mu;
  red[t] = d0 * d0 + d1 * d1 + d2 * d2 + d3 * d3;
  __syncthreads();
  for (int o = 128; o > 0; o >>= 1) {
    if (t < o) red[t] += red[t + o];
    __syncthreads();
  }
  const float rs = rsqrtf(red[0] * (1.0f / C) + 1e-3f);

  const float4 gv = ((const float4*)g)[t];
  const float4 bv = ((const float4*)bta)[t];
  _Float16* o4 = out + row * C + t * 4;
  o4[0] = (_Float16)(d0 * rs * gv.x + bv.x);
  o4[1] = (_Float16)(d1 * rs * gv.y + bv.y);
  o4[2] = (_Float16)(d2 * rs * gv.z + bv.z);
  o4[3] = (_Float16)(d3 * rs * gv.w + bv.w);
}

// ---------------------------------------------------------------------------
// WMMA GEMM: out[M,N] = A[M,K]*B[K,N] (f16 in, f32 accum), 64x64 tile,
// 256 thr = 8 waves x 2 subtiles. Tiles staged into LDS in fragment order.
// ---------------------------------------------------------------------------
template <typename OutT, bool BIAS, bool RELU, bool RES>
__global__ __launch_bounds__(256) void gemm_wmma(
    const _Float16* __restrict__ A, const _Float16* __restrict__ B,
    const float* __restrict__ bias, const float* __restrict__ res,
    OutT* __restrict__ out, int M, int N, int K) {
  __shared__ __align__(32) _Float16 As[4 * 512];  // [mi][lane][16]
  __shared__ __align__(32) _Float16 Bs[4 * 512];  // [ni][lane][16]

  const int t = threadIdx.x;
  const int lane = t & 31;
  const int wave = t >> 5;
  const int bn = blockIdx.x * 64;
  const int bm = blockIdx.y * 64;
  const int mi = (wave * 2) >> 2;
  const int ni = (wave * 2) & 3;

  // A staging: thread owns 8 contiguous halves (row ra, cols ca..ca+7);
  // swizzle is contiguity-preserving -> single 16B store / async copy.
  const int ra = (t * 8) >> 5, ca = (t * 8) & 31;
  const int aslot =
      (ra >> 4) * 512 + ((ra & 15) + 16 * ((ca >> 3) & 1)) * 16 + 8 * (ca >> 4);
  const size_t agb = (size_t)(bm + ra) * K + ca;
  // B staging: 8 contiguous halves (row rb=k, cols cb..cb+7) scatter with
  // stride 16 halves (adjacent n -> adjacent lane slot).
  const int rb = (t * 8) >> 6, cb = (t * 8) & 63;
  const int bslot =
      (cb >> 4) * 512 + ((cb & 15) + 16 * ((rb >> 3) & 1)) * 16 + (rb & 7) + 8 * (rb >> 4);
  const size_t bgb = (size_t)rb * N + bn + cb;

  v8f acc0 = {};
  v8f acc1 = {};

  for (int k0 = 0; k0 < K; k0 += 32) {
    stage16B(&A[agb + k0], &As[aslot]);
    {
      const v8h hv = *(const v8h*)&B[bgb + (size_t)k0 * N];
#pragma unroll
      for (int e = 0; e < 8; ++e) Bs[bslot + e * 16] = hv[e];
    }
    if (k0 + 32 < K) {
      __builtin_prefetch(&A[agb + k0 + 32], 0, 0);
      __builtin_prefetch(&B[bgb + (size_t)(k0 + 32) * N], 0, 0);
    }
    async_fence();
    __syncthreads();

    const v16h a  = ld_frag(As, mi, lane);
    const v16h b0 = ld_frag(Bs, ni, lane);
    const v16h b1 = ld_frag(Bs, ni + 1, lane);
    acc0 = wmma_f16(a, b0, acc0);
    acc1 = wmma_f16(a, b1, acc1);
    __syncthreads();
  }

  const int n = lane & 15;
  const int mb = (lane >= 16) ? 8 : 0;
#pragma unroll
  for (int j = 0; j < 8; ++j) {
    const int gm = bm + mi * 16 + mb + j;
    {
      const int gn = bn + ni * 16 + n;
      float v = acc0[j];
      if (BIAS) v += bias[gn];
      if (RELU) v = fmaxf(v, 0.0f);
      if (RES) v += res[(size_t)gm * N + gn];
      out[(size_t)gm * N + gn] = (OutT)v;
    }
    {
      const int gn = bn + (ni + 1) * 16 + n;
      float v = acc1[j];
      if (BIAS) v += bias[gn];
      if (RELU) v = fmaxf(v, 0.0f);
      if (RES) v += res[(size_t)gm * N + gn];
      out[(size_t)gm * N + gn] = (OutT)v;
    }
  }
}

// ---------------------------------------------------------------------------
// Causal flash attention, Dk=64, 64-query tile per block, 256 threads.
// All matrix tiles held in LDS in WMMA fragment order. Scores MULTIPLIED
// by sqrt(Dk)=8 (source quirk).
// ---------------------------------------------------------------------------
__global__ __launch_bounds__(256) void attn_flash(
    const _Float16* __restrict__ Q, const _Float16* __restrict__ Kb,
    const _Float16* __restrict__ Vb, _Float16* __restrict__ ctx) {
  constexpr int T = 2048, C = 1024;
  __shared__ __align__(32) _Float16 Qs[8 * 512];  // A-layout [mi*2+kc][lane][16]
  __shared__ __align__(32) _Float16 Ks[8 * 512];  // B-layout for Q*K^T
  __shared__ __align__(32) _Float16 Vs[8 * 512];  // B-layout for P*V
  __shared__ __align__(32) _Float16 Ps[8 * 512];  // A-layout
  __shared__ __align__(32) float Ss[64 * 64];     // C-layout (swizzled)
  __shared__ __align__(32) float Os[64 * 64];     // C-layout (swizzled)
  __shared__ float mrow[64], lrow[64], crow[64];

  const int t = threadIdx.x;
  const int lane = t & 31;
  const int wave = t >> 5;
  const int qt = blockIdx.x & 31;
  const int h = (blockIdx.x >> 5) & 15;
  const int b = blockIdx.x >> 9;
  const int q0 = qt * 64;
  const size_t base = (size_t)b * T * C + (size_t)h * 64;

  const int mi = (wave * 2) >> 2;
  const int ni = (wave * 2) & 3;

  // Per-thread staging geometry: 16 contiguous halves (row r, cols c..c+15).
  const int r = (t * 16) >> 6;
  const int c = (t * 16) & 63;
  // Q (A-operand): splits into two 8-half runs (lane-halves of the K group).
  const int qsub = ((r >> 4) * 2 + (c >> 5)) * 512 + 8 * ((c & 31) >> 4);
  const int qs0 = qsub + (r & 15) * 16;
  const int qs1 = qsub + ((r & 15) + 16) * 16;
  // K (transposed-B operand): ksl is unused; K is staged at qs0/qs1, same layout as Q.
  const int ksl =
      ((r >> 4) * 2 + (c >> 5)) * 512 + ((r & 15) + 16 * ((c & 31) >> 4)) * 16;
  // V (B-operand): scatter, stride 16 halves per feature.
  const int vsl =
      ((c >> 4) * 2 + (r >> 5)) * 512 + 16 * 16 * ((r >> 3) & 1) + (r & 7) + 8 * ((r & 31) >> 4);

  {  // load Q tile once
    const _Float16* src = &Q[base + (size_t)(q0 + r) * C + c];
    stage16B(src, &Qs[qs0]);
    stage16B(src + 8, &Qs[qs1]);
  }
  for (int i = t; i < 64 * 64; i += 256) Os[i] = 0.0f;
  if (t < 64) {
    mrow[t] = -1e30f;
    lrow[t] = 0.0f;
  }
  async_fence();
  __syncthreads();

  for (int kt = 0; kt <= qt; ++kt) {
    {  // stage K (two 8-half runs, like Q) and V (repack through VGPRs)
      const size_t g = base + (size_t)(kt * 64 + r) * C + c;
      stage16B(&Kb[g], &Ks[qs0]);
      stage16B(&Kb[g] + 8, &Ks[qs1]);
      const v16h hv = *(const v16h*)&Vb[g];
#pragma unroll
      for (int e = 0; e < 16; ++e) Vs[vsl + e * 16] = hv[e];
      if (kt < qt) {
        __builtin_prefetch(&Kb[g + (size_t)64 * C], 0, 0);
        __builtin_prefetch(&Vb[g + (size_t)64 * C], 0, 0);
      }
    }
    async_fence();
    __syncthreads();

    {  // S = (Q · K^T) * 8
      v8f s0 = {};
      v8f s1 = {};
#pragma unroll
      for (int kc = 0; kc < 2; ++kc) {
        const v16h a = ld_frag(Qs, mi * 2 + kc, lane);
        const v16h b0 = ld_frag(Ks, ni * 2 + kc, lane);
        const v16h b1 = ld_frag(Ks, (ni + 1) * 2 + kc, lane);
        s0 = wmma_f16(a, b0, s0);
        s1 = wmma_f16(a, b1, s1);
      }
      *(v8f*)&Ss[((mi * 4 + ni) * 32 + lane) * 8] = s0 * 8.0f;
      *(v8f*)&Ss[((mi * 4 + ni + 1) * 32 + lane) * 8] = s1 * 8.0f;
    }
    __syncthreads();

    if (t < 64) {  // online softmax bookkeeping, one row per thread
      const int gq = q0 + t;
      const float m_old = mrow[t];
      float tm = -1e30f;
#pragma unroll 8
      for (int n = 0; n < 64; ++n)
        if (kt * 64 + n <= gq) tm = fmaxf(tm, Ss[cslot(t, n)]);
      const float m_new = fmaxf(m_old, tm);
      const float corr = __expf(m_old - m_new);
      float ps = 0.0f;
#pragma unroll 8
      for (int n = 0; n < 64; ++n) {
        float p = 0.0f;
        if (kt * 64 + n <= gq) {
          p = __expf(Ss[cslot(t, n)] - m_new);
          ps += p;
        }
        Ps[aslot64(t, n)] = (_Float16)p;  // write P directly in A-layout
      }
      lrow[t] = lrow[t] * corr + ps;
      mrow[t] = m_new;
      crow[t] = corr;
    }
    __syncthreads();

    for (int i = t; i < 64 * 64; i += 256) {  // rescale O by row correction
      const int rr = ((i >> 8) >> 2) * 16 + (((i >> 3) & 31) >> 4) * 8 + (i & 7);
      Os[i] *= crow[rr];
    }
    __syncthreads();

    {  // O += P · V
      const int o0 = ((mi * 4 + ni) * 32 + lane) * 8;
      const int o1 = ((mi * 4 + ni + 1) * 32 + lane) * 8;
      v8f c0 = *(const v8f*)&Os[o0];
      v8f c1 = *(const v8f*)&Os[o1];
#pragma unroll
      for (int kc = 0; kc < 2; ++kc) {
        const v16h a = ld_frag(Ps, mi * 2 + kc, lane);
        const v16h b0 = ld_frag(Vs, ni * 2 + kc, lane);
        const v16h b1 = ld_frag(Vs, (ni + 1) * 2 + kc, lane);
        c0 = wmma_f16(a, b0, c0);
        c1 = wmma_f16(a, b1, c1);
      }
      *(v8f*)&Os[o0] = c0;
      *(v8f*)&Os[o1] = c1;
    }
    __syncthreads();
  }

  for (int i = t; i < 64 * 64; i += 256) {  // ctx = O / l
    const int pair = i >> 8, lane_ = (i >> 3) & 31, j = i & 7;
    const int rr = (pair >> 2) * 16 + (lane_ >> 4) * 8 + j;
    const int cc = (pair & 3) * 16 + (lane_ & 15);
    ctx[base + (size_t)(q0 + rr) * C + cc] = (_Float16)(Os[i] / lrow[rr]);
  }
}

// ---------------------------------------------------------------------------
// Host launcher
// ---------------------------------------------------------------------------
extern "C" void kernel_launch(void* const* d_in, const int* in_sizes, int n_in,
                              void* d_out, int out_size, void* d_ws,
                              size_t ws_size, hipStream_t stream) {
  (void)in_sizes; (void)n_in; (void)out_size; (void)ws_size;
  constexpr int B = 2, T = 2048, C = 1024, INNER = 512;
  constexpr int M = B * T;  // 4096
  const size_t MB = 1u << 20;

  const float* x = (const float*)d_in[0];
  const float* q_w = (const float*)d_in[1];
  const float* k_w = (const float*)d_in[2];
  const float* v_w = (const float*)d_in[3];
  const float* o_w = (const float*)d_in[4];
  const float* ln_g = (const float*)d_in[5];
  const float* ln_b = (const float*)d_in[6];
  const float* w1 = (const float*)d_in[7];
  const float* b1 = (const float*)d_in[8];
  const float* w2 = (const float*)d_in[9];
  const float* b2 = (const float*)d_in[10];
  float* out = (float*)d_out;

  char* ws = (char*)d_ws;
  _Float16* hF   = (_Float16*)(ws + 0 * MB);
  _Float16* qF   = (_Float16*)(ws + 8 * MB);
  _Float16* kF   = (_Float16*)(ws + 16 * MB);
  _Float16* vF   = (_Float16*)(ws + 24 * MB);
  _Float16* ctxF = (_Float16*)(ws + 32 * MB);
  float*    x1   = (float*)(ws + 40 * MB);
  _Float16* h2F  = (_Float16*)(ws + 56 * MB);
  _Float16* f1F  = (_Float16*)(ws + 64 * MB);
  _Float16* qwF  = (_Float16*)(ws + 68 * MB);
  _Float16* kwF  = (_Float16*)(ws + 70 * MB);
  _Float16* vwF  = (_Float16*)(ws + 72 * MB);
  _Float16* owF  = (_Float16*)(ws + 74 * MB);
  _Float16* w1F  = (_Float16*)(ws + 76 * MB);
  _Float16* w2F  = (_Float16*)(ws + 77 * MB);

  const int WCC = C * C, WCI = C * INNER;
  cvt_f32_to_f16<<<(WCC + 255) / 256, 256, 0, stream>>>(q_w, qwF, WCC);
  cvt_f32_to_f16<<<(WCC + 255) / 256, 256, 0, stream>>>(k_w, kwF, WCC);
  cvt_f32_to_f16<<<(WCC + 255) / 256, 256, 0, stream>>>(v_w, vwF, WCC);
  cvt_f32_to_f16<<<(WCC + 255) / 256, 256, 0, stream>>>(o_w, owF, WCC);
  cvt_f32_to_f16<<<(WCI + 255) / 256, 256, 0, stream>>>(w1, w1F, WCI);
  cvt_f32_to_f16<<<(WCI + 255) / 256, 256, 0, stream>>>(w2, w2F, WCI);

  ln_f16<<<M, 256, 0, stream>>>(x, ln_g, ln_b, hF);

  dim3 gCC(C / 64, M / 64);
  gemm_wmma<_Float16, false, false, false>
      <<<gCC, 256, 0, stream>>>(hF, qwF, nullptr, nullptr, qF, M, C, C);
  gemm_wmma<_Float16, false, false, false>
      <<<gCC, 256, 0, stream>>>(hF, kwF, nullptr, nullptr, kF, M, C, C);
  gemm_wmma<_Float16, false, false, false>
      <<<gCC, 256, 0, stream>>>(hF, vwF, nullptr, nullptr, vF, M, C, C);

  attn_flash<<<B * 16 * (T / 64), 256, 0, stream>>>(qF, kF, vF, ctxF);

  gemm_wmma<float, false, false, true>
      <<<gCC, 256, 0, stream>>>(ctxF, owF, nullptr, x, x1, M, C, C);

  ln_f16<<<M, 256, 0, stream>>>(x1, ln_g, ln_b, h2F);

  dim3 gCI(INNER / 64, M / 64);
  gemm_wmma<_Float16, true, true, false>
      <<<gCI, 256, 0, stream>>>(h2F, w1F, b1, nullptr, f1F, M, INNER, C);

  gemm_wmma<float, true, false, true>
      <<<gCC, 256, 0, stream>>>(f1F, w2F, b2, x1, out, M, C, INNER);
}
